// MultiHeadAttention_82059645157715
// MI455X (gfx1250) — matrix operand order repaired and checked by reading, not compiled, not run on hardware
//
#include <hip/hip_runtime.h>

typedef _Float16 v8h  __attribute__((ext_vector_type(8)));
typedef _Float16 v16h __attribute__((ext_vector_type(16)));
typedef float    v8f  __attribute__((ext_vector_type(8)));

#define WMMA_F16(A, B, C) \
  __builtin_amdgcn_wmma_f32_16x16x32_f16(false, (A), false, (B), (short)0, (C), false, false)

#define SEQ    2048
#define DMODEL 1024
#define NHEAD  16
#define DHEAD  64

__device__ __forceinline__ v16h ld_frag16(const _Float16* p0, const _Float16* p1) {
  v8h lo = *(const v8h*)p0;
  v8h hi = *(const v8h*)p1;
  v16h r;
#pragma unroll
  for (int i = 0; i < 8; ++i) { r[i] = lo[i]; r[i + 8] = hi[i]; }
  return r;
}

// LDS byte offset of a generic pointer to a __shared__ object: flat LDS
// addresses carry the LDS offset in addr[31:0] (aperture mapping truncates).
__device__ __forceinline__ unsigned lds_off(const void* p) {
  return (unsigned)(unsigned long long)(uintptr_t)p;
}

// Async DMA: 16 bytes per lane, global -> LDS, tracked by ASYNCcnt.
__device__ __forceinline__ void async_ld16(unsigned ldsBytes, const void* g) {
  unsigned long long ga = (unsigned long long)(uintptr_t)g;
  asm volatile("global_load_async_to_lds_b128 %0, %1, off"
               :: "v"(ldsBytes), "v"(ga) : "memory");
}
__device__ __forceinline__ void wait_async0() {
  asm volatile("s_wait_asynccnt 0" ::: "memory");
}

// ---------------------------------------------------------------------------
// QKV projection: out = (X[8192x1024] @ W[1024x1024] + bias) * scale, cast f16
// kt_layout==0: out[((b*16+h)*2048+s)*64+d]   (row-major per head, for Q and V)
// kt_layout==1: out[((b*16+h)*64+d)*2048+s]   (transposed, for K)
// ---------------------------------------------------------------------------
__global__ __launch_bounds__(256) void proj_kernel(
    const float* __restrict__ X, const float* __restrict__ W,
    const float* __restrict__ bias, _Float16* __restrict__ out,
    float scale, int kt_layout)
{
  __shared__ __attribute__((aligned(16))) _Float16 As[64 * 40];   // stride 40 halves (80B)
  __shared__ __attribute__((aligned(16))) _Float16 Bs[32 * 136];  // stride 136 halves (272B)

  const int tid  = threadIdx.x;
  const int lane = tid & 31;
  const int wid  = tid >> 5;
  const int bm = (blockIdx.x >> 3) * 64;   // 128 M-tiles
  const int bn = (blockIdx.x & 7) * 128;   // 8 N-tiles
  const int wm = (wid >> 2) * 32;
  const int wn = (wid & 3) * 32;
  const int mm = lane & 15;
  const int hh = lane >> 4;

  v8f acc[2][2] = {};

  const int arow = tid >> 2, acol = (tid & 3) * 8;   // A tile copy: 8 elems/thread
  const int brow = tid >> 3, bcol = (tid & 7) * 16;  // B tile copy: 16 elems/thread
  const float* aSrc = X + (size_t)(bm + arow) * DMODEL + acol;
  const float* bSrc = W + (size_t)brow * DMODEL + bn + bcol;

  for (int k0 = 0; k0 < DMODEL; k0 += 32) {
    __syncthreads();
    if (k0 + 32 < DMODEL) {  // prefetch next k-slice (global_prefetch_b8)
      __builtin_prefetch(aSrc + k0 + 32, 0, 1);
      __builtin_prefetch(bSrc + (size_t)(k0 + 32) * DMODEL, 0, 1);
    }
    {
      const float4* src = (const float4*)(aSrc + k0);
      float4 x0 = src[0], x1 = src[1];
      _Float16* d = &As[arow * 40 + acol];
      d[0]=(_Float16)x0.x; d[1]=(_Float16)x0.y; d[2]=(_Float16)x0.z; d[3]=(_Float16)x0.w;
      d[4]=(_Float16)x1.x; d[5]=(_Float16)x1.y; d[6]=(_Float16)x1.z; d[7]=(_Float16)x1.w;
    }
    {
      const float4* src = (const float4*)(bSrc + (size_t)k0 * DMODEL);
      float4 x0 = src[0], x1 = src[1], x2 = src[2], x3 = src[3];
      _Float16* d = &Bs[brow * 136 + bcol];
      d[0] =(_Float16)x0.x; d[1] =(_Float16)x0.y; d[2] =(_Float16)x0.z; d[3] =(_Float16)x0.w;
      d[4] =(_Float16)x1.x; d[5] =(_Float16)x1.y; d[6] =(_Float16)x1.z; d[7] =(_Float16)x1.w;
      d[8] =(_Float16)x2.x; d[9] =(_Float16)x2.y; d[10]=(_Float16)x2.z; d[11]=(_Float16)x2.w;
      d[12]=(_Float16)x3.x; d[13]=(_Float16)x3.y; d[14]=(_Float16)x3.z; d[15]=(_Float16)x3.w;
    }
    __syncthreads();

    v16h a[2], b[2];
#pragma unroll
    for (int tm = 0; tm < 2; ++tm) {
      const _Float16* base = &As[(wm + tm * 16 + mm) * 40];
      a[tm] = ld_frag16(base + 8 * hh, base + 16 + 8 * hh);
    }
#pragma unroll
    for (int tn = 0; tn < 2; ++tn) {
      const _Float16* base = &Bs[wn + tn * 16 + mm];
#pragma unroll
      for (int s = 0; s < 8; ++s) {
        b[tn][s]     = base[(8 * hh + s) * 136];
        b[tn][s + 8] = base[(16 + 8 * hh + s) * 136];
      }
    }
#pragma unroll
    for (int tm = 0; tm < 2; ++tm)
#pragma unroll
      for (int tn = 0; tn < 2; ++tn)
        acc[tm][tn] = WMMA_F16(a[tm], b[tn], acc[tm][tn]);
  }

#pragma unroll
  for (int tm = 0; tm < 2; ++tm) {
#pragma unroll
    for (int tn = 0; tn < 2; ++tn) {
      const int gn = bn + wn + tn * 16 + mm;
      const float bv = bias[gn];
      const int hIdx = gn >> 6, dIdx = gn & 63;
#pragma unroll
      for (int r = 0; r < 8; ++r) {
        const int gm = bm + wm + tm * 16 + r + 8 * hh;   // = b*2048+s
        const int bb = gm >> 11, ss = gm & 2047;
        const float val = (acc[tm][tn][r] + bv) * scale;
        size_t off = kt_layout
            ? (((size_t)(bb * NHEAD + hIdx) * DHEAD + dIdx) * SEQ + ss)
            : (((size_t)(bb * NHEAD + hIdx) * SEQ + ss) * DHEAD + dIdx);
        out[off] = (_Float16)val;
      }
    }
  }
}

// ---------------------------------------------------------------------------
// Flash attention: one WG = one (b,h) x 128-query block; wave = 16 query rows.
// K/V tiles double-buffered in LDS, filled by async global->LDS DMA that
// overlaps the WMMA + softmax of the previous tile.
// ---------------------------------------------------------------------------
__global__ __launch_bounds__(256) void attn_kernel(
    const _Float16* __restrict__ Q, const _Float16* __restrict__ Kt,
    const _Float16* __restrict__ V, float* __restrict__ out)
{
  __shared__ __attribute__((aligned(16))) _Float16 Ks[2][64 * 40];   // [dh][key], stride 40
  __shared__ __attribute__((aligned(16))) _Float16 Vs[2][32 * 72];   // [key][dh], stride 72
  __shared__ __attribute__((aligned(16))) _Float16 Ps[8][16 * 40];   // per-wave P

  const int tid = threadIdx.x, lane = tid & 31, wid = tid >> 5;
  const int bh   = blockIdx.x >> 4;
  const int qblk = blockIdx.x & 15;
  const int mm = lane & 15, hh = lane >> 4;
  const int qrow0 = qblk * 128 + wid * 16;

  // Q fragments (16 rows x 64 dh -> two 16x32 A-frags), straight from global.
  const _Float16* qrow = Q + ((size_t)bh * SEQ + qrow0 + mm) * DHEAD;
  const v16h aq0 = ld_frag16(qrow + 8 * hh,      qrow + 16 + 8 * hh);
  const v16h aq1 = ld_frag16(qrow + 32 + 8 * hh, qrow + 48 + 8 * hh);

  v8f o0 = {}, o1 = {}, o2 = {}, o3 = {};
  float mrow[8], lrow[8];
#pragma unroll
  for (int r = 0; r < 8; ++r) { mrow[r] = -1e30f; lrow[r] = 0.0f; }

  // Per-thread 16B copy slots for the K (64x32) and V (32x64) tiles.
  const int kdh = tid >> 2,  kc = (tid & 3) * 8;
  const int vky = tid >> 3,  vc = (tid & 7) * 8;
  const unsigned ksOff[2] = { lds_off(&Ks[0][kdh * 40 + kc]), lds_off(&Ks[1][kdh * 40 + kc]) };
  const unsigned vsOff[2] = { lds_off(&Vs[0][vky * 72 + vc]), lds_off(&Vs[1][vky * 72 + vc]) };
  const _Float16* ktSrc = Kt + ((size_t)bh * DHEAD + kdh) * SEQ + kc;   // + j
  const _Float16* vSrc  = V  + ((size_t)bh * SEQ + vky) * DHEAD + vc;   // + j*DHEAD

  // One 32-key tile: 4 WMMA for scores, online softmax, 4 WMMA for P*V.
  auto compute_tile = [&](const _Float16* KsBuf, const _Float16* VsBuf) {
    v16h bk00, bk10, bk01, bk11;
#pragma unroll
    for (int s = 0; s < 8; ++s) {
      const int ka = 8 * hh + s, kb = 16 + 8 * hh + s;
      bk00[s] = KsBuf[ka * 40 + mm];             bk00[s + 8] = KsBuf[kb * 40 + mm];
      bk10[s] = KsBuf[(32 + ka) * 40 + mm];      bk10[s + 8] = KsBuf[(32 + kb) * 40 + mm];
      bk01[s] = KsBuf[ka * 40 + 16 + mm];        bk01[s + 8] = KsBuf[kb * 40 + 16 + mm];
      bk11[s] = KsBuf[(32 + ka) * 40 + 16 + mm]; bk11[s + 8] = KsBuf[(32 + kb) * 40 + 16 + mm];
    }
    v8f s0 = {}, s1 = {};
    s0 = WMMA_F16(aq0, bk00, s0);
    s0 = WMMA_F16(aq1, bk10, s0);
    s1 = WMMA_F16(aq0, bk01, s1);
    s1 = WMMA_F16(aq1, bk11, s1);

    float p0[8], p1[8], corr[8];
#pragma unroll
    for (int r = 0; r < 8; ++r) {
      float mx = fmaxf(s0[r], s1[r]);
      mx = fmaxf(mx, __shfl_xor(mx, 1, 32));
      mx = fmaxf(mx, __shfl_xor(mx, 2, 32));
      mx = fmaxf(mx, __shfl_xor(mx, 4, 32));
      mx = fmaxf(mx, __shfl_xor(mx, 8, 32));
      const float mn = fmaxf(mrow[r], mx);
      corr[r] = __builtin_exp2f(mrow[r] - mn);
      p0[r] = __builtin_exp2f(s0[r] - mn);
      p1[r] = __builtin_exp2f(s1[r] - mn);
      float rs = p0[r] + p1[r];
      rs += __shfl_xor(rs, 1, 32);
      rs += __shfl_xor(rs, 2, 32);
      rs += __shfl_xor(rs, 4, 32);
      rs += __shfl_xor(rs, 8, 32);
      lrow[r] = lrow[r] * corr[r] + rs;
      mrow[r] = mn;
    }
#pragma unroll
    for (int r = 0; r < 8; ++r) {
      o0[r] *= corr[r]; o1[r] *= corr[r]; o2[r] *= corr[r]; o3[r] *= corr[r];
    }

    // C-layout -> A-layout for P via per-wave LDS buffer (LDS is in-order per wave).
    _Float16* pw = Ps[wid];
#pragma unroll
    for (int r = 0; r < 8; ++r) {
      const int row = r + 8 * hh;
      pw[row * 40 + mm]      = (_Float16)p0[r];
      pw[row * 40 + 16 + mm] = (_Float16)p1[r];
    }
    asm volatile("s_wait_dscnt 0" ::: "memory");
    const v16h ap = ld_frag16(&pw[mm * 40 + 8 * hh], &pw[mm * 40 + 16 + 8 * hh]);

    v16h bv0, bv1, bv2, bv3;
#pragma unroll
    for (int s = 0; s < 8; ++s) {
      const _Float16* va = &VsBuf[(8 * hh + s) * 72 + mm];
      const _Float16* vb = &VsBuf[(16 + 8 * hh + s) * 72 + mm];
      bv0[s] = va[0];  bv0[s + 8] = vb[0];
      bv1[s] = va[16]; bv1[s + 8] = vb[16];
      bv2[s] = va[32]; bv2[s + 8] = vb[32];
      bv3[s] = va[48]; bv3[s + 8] = vb[48];
    }
    o0 = WMMA_F16(ap, bv0, o0);
    o1 = WMMA_F16(ap, bv1, o1);
    o2 = WMMA_F16(ap, bv2, o2);
    o3 = WMMA_F16(ap, bv3, o3);
  };

  // Prologue: DMA tile 0 into buffer 0.
  async_ld16(ksOff[0], ktSrc);
  async_ld16(vsOff[0], vSrc);
  wait_async0();
  __syncthreads();

  for (int j = 0; j < SEQ; j += 64) {
    // Prefetch tile (j+32) into buffer 1, compute tile j from buffer 0.
    async_ld16(ksOff[1], ktSrc + (j + 32));
    async_ld16(vsOff[1], vSrc + (size_t)(j + 32) * DHEAD);
    compute_tile(Ks[0], Vs[0]);
    wait_async0();
    __syncthreads();

    // Prefetch tile (j+64) into buffer 0, compute tile j+32 from buffer 1.
    if (j + 64 < SEQ) {
      async_ld16(ksOff[0], ktSrc + (j + 64));
      async_ld16(vsOff[0], vSrc + (size_t)(j + 64) * DHEAD);
    }
    compute_tile(Ks[1], Vs[1]);
    wait_async0();
    __syncthreads();
  }

  // Normalize and store fp32 output in [B, S, H*Dh].
  const int b = bh >> 4, hIdx = bh & 15;
#pragma unroll
  for (int r = 0; r < 8; ++r) {
    const float inv = 1.0f / lrow[r];
    const int ss = qrow0 + r + 8 * hh;
    float* orow = out + (((size_t)b * SEQ + ss) * NHEAD + hIdx) * DHEAD + mm;
    orow[0]  = o0[r] * inv;
    orow[16] = o1[r] * inv;
    orow[32] = o2[r] * inv;
    orow[48] = o3[r] * inv;
  }
}

// ---------------------------------------------------------------------------
extern "C" void kernel_launch(void* const* d_in, const int* in_sizes, int n_in,
                              void* d_out, int out_size, void* d_ws, size_t ws_size,
                              hipStream_t stream) {
  const float* key   = (const float*)d_in[0];
  const float* value = (const float*)d_in[1];
  const float* query = (const float*)d_in[2];
  const float* Wq = (const float*)d_in[3];
  const float* bq = (const float*)d_in[4];
  const float* Wk = (const float*)d_in[5];
  const float* bk = (const float*)d_in[6];
  const float* Wv = (const float*)d_in[7];
  const float* bv = (const float*)d_in[8];
  float* out = (float*)d_out;

  const size_t perTensor = (size_t)4 * NHEAD * SEQ * DHEAD;  // 8M f16 elems
  _Float16* qh = (_Float16*)d_ws;
  _Float16* kt = qh + perTensor;
  _Float16* vh = kt + perTensor;

  const float qscale = 1.4426950408889634f / 8.0f;  // log2(e)/sqrt(Dh)

  dim3 blk(256);
  proj_kernel<<<dim3(1024), blk, 0, stream>>>(query, Wq, bq, qh, qscale, 0);
  proj_kernel<<<dim3(1024), blk, 0, stream>>>(key,   Wk, bk, kt, 1.0f,   1);
  proj_kernel<<<dim3(1024), blk, 0, stream>>>(value, Wv, bv, vh, 1.0f,   0);
  attn_kernel<<<dim3(1024), blk, 0, stream>>>(qh, kt, vh, out);
}
